// NeuroManifoldGPT_12618613915820
// MI455X (gfx1250) — matrix operand order repaired and checked by reading, not compiled, not run on hardware
//
#include <hip/hip_runtime.h>
#include <math.h>

// ---------------------------------------------------------------------------
// NeuroManifoldGPT forward for MI455X (gfx1250, wave32, WMMA bf16 path).
// All GEMMs: f32 in HBM, converted to bf16 (v_cvt_pk_bf16_f32) while staging
// tiles into double-buffered LDS, accumulated in f32 via
// v_wmma_f32_16x16x32_bf16. Block tile 128x128, wave tile 32x64
// (2x4 WMMA 16x16 tiles), K-step 32, register-staged global loads.
// ---------------------------------------------------------------------------

typedef __attribute__((ext_vector_type(16))) __bf16 v16bf;
typedef __attribute__((ext_vector_type(8)))  float  v8f;
typedef __attribute__((ext_vector_type(8)))  __bf16 bf16x8;
typedef __attribute__((ext_vector_type(4)))  __bf16 bf16x4;

#define NTOK  2048      // B*T
#define VOCAB 32000
#define SDIM  2048
#define KACT  40
#define EDIM  256
#define EMB   768
#define NLAYER 4
#define HID   3072
#define CAP   65536

// Hardware f32 -> bf16 convert (lowers to v_cvt_pk_bf16_f32 when paired)
__device__ __forceinline__ __bf16 f2bf(float f) { return (__bf16)f; }

union FragU { bf16x8 h[2]; v16bf v; };

// ---------------------------------------------------------------------------
// Generic GEMM: C[M,N] = A[M,K] @ B[K,N] (+R1)(+R2)(relu?), all f32 row-major.
// 256 threads = 8 waves. Waves: 4 along M (32 rows each) x 2 along N (64 each).
// ---------------------------------------------------------------------------
#define TLD 40   // LDS row stride (bf16 elems): 80B, 16B-aligned, bank-skewed

__global__ __launch_bounds__(256) void gemm_bf16_wmma(
    const float* __restrict__ A, int lda,
    const float* __restrict__ B, int ldb,
    float* __restrict__ C, int ldc,
    const float* __restrict__ R1,
    const float* __restrict__ R2,
    int Kd, int relu)
{
    // double-buffered tiles: A row-major [m][k], B transposed [n][k]
    __shared__ __align__(16) __bf16 Al[2][128][TLD];
    __shared__ __align__(16) __bf16 Bt[2][128][TLD];

    const int tid  = threadIdx.x;
    const int lane = tid & 31;
    const int wave = tid >> 5;
    const int wm   = (wave >> 1) * 32;   // wave M offset (0..96)
    const int wn   = (wave & 1) * 64;    // wave N offset (0/64)
    const int bm   = blockIdx.y * 128;
    const int bn   = blockIdx.x * 128;

    // A staging: thread -> 1 row x 16 cols
    const int ar  = tid >> 1;            // A row 0..127
    const int ac  = (tid & 1) * 16;      // A col base {0,16}
    // B staging: thread -> 4(k) x 4(n) sub-block (k-contiguous LDS stores)
    const int kb4 = (tid >> 5) * 4;      // k base 0..28
    const int nb4 = (tid & 31) * 4;      // n base 0..124

    v8f zero = {};
    v8f acc[2][4];
    #pragma unroll
    for (int i = 0; i < 2; ++i)
        #pragma unroll
        for (int j = 0; j < 4; ++j) acc[i][j] = zero;

    // fragment lane mapping (ISA 7.12.2, 16-bit A 16x32 / B 32x16)
    const int lm  = lane & 15;
    const int kbA = (lane < 16) ? 0 : 8;   // A: elems 0..7 = K kbA.., 8..15 = K 16+kbA..
    const int kbB = (lane < 16) ? 0 : 8;   // B: elems 0..7 = K kbB.., 8..15 = K 16+kbB..

    float4 ra[4];   // staged A: 16 floats (1 row x 16 k)
    float4 rb[4];   // staged B: rb[r] = 4 n-values of k-row r

    auto stage_load = [&](int k0) {
        const float4* ag = (const float4*)(A + (size_t)(bm + ar) * lda + k0 + ac);
        #pragma unroll
        for (int q = 0; q < 4; ++q) ra[q] = ag[q];
        #pragma unroll
        for (int r = 0; r < 4; ++r)
            rb[r] = *(const float4*)(B + (size_t)(k0 + kb4 + r) * ldb + bn + nb4);
    };

    auto stage_store = [&](int b) {
        const float* af = (const float*)ra;
        bf16x8 p0, p1;
        #pragma unroll
        for (int j = 0; j < 8; ++j) { p0[j] = f2bf(af[j]); p1[j] = f2bf(af[8 + j]); }
        __bf16* ad = &Al[b][ar][ac];
        *(bf16x8*)ad = p0;
        *(bf16x8*)(ad + 8) = p1;
        #pragma unroll
        for (int j = 0; j < 4; ++j) {
            bf16x4 p;
            p[0] = f2bf(((const float*)&rb[0])[j]);
            p[1] = f2bf(((const float*)&rb[1])[j]);
            p[2] = f2bf(((const float*)&rb[2])[j]);
            p[3] = f2bf(((const float*)&rb[3])[j]);
            *(bf16x4*)&Bt[b][nb4 + j][kb4] = p;
        }
    };

    // prologue: stage first K-tile
    stage_load(0);
    stage_store(0);
    __syncthreads();

    int buf = 0;
    for (int k0 = 0; k0 < Kd; k0 += 32) {
        const bool more = (k0 + 32) < Kd;
        if (more) stage_load(k0 + 32);   // global loads in flight during WMMAs

        FragU af[2], bfm[4];
        #pragma unroll
        for (int mt = 0; mt < 2; ++mt) {
            const __bf16* p = &Al[buf][wm + mt*16 + lm][0];
            af[mt].h[0] = *(const bf16x8*)(p + kbA);
            af[mt].h[1] = *(const bf16x8*)(p + kbA + 16);
        }
        #pragma unroll
        for (int nt = 0; nt < 4; ++nt) {
            const __bf16* p = &Bt[buf][wn + nt*16 + lm][0];
            bfm[nt].h[0] = *(const bf16x8*)(p + kbB);
            bfm[nt].h[1] = *(const bf16x8*)(p + kbB + 16);
        }
        #pragma unroll
        for (int mt = 0; mt < 2; ++mt)
            #pragma unroll
            for (int nt = 0; nt < 4; ++nt)
                acc[mt][nt] = __builtin_amdgcn_wmma_f32_16x16x32_bf16(
                    false, af[mt].v, false, bfm[nt].v,
                    (short)0, acc[mt][nt], false, false);

        if (more) stage_store(buf ^ 1);  // cvt + LDS store after compute
        __syncthreads();
        buf ^= 1;
    }

    // ---- epilogue: fused residual adds + relu, f32 store ----
    #pragma unroll
    for (int mt = 0; mt < 2; ++mt)
    #pragma unroll
    for (int nt = 0; nt < 4; ++nt)
    #pragma unroll
    for (int r = 0; r < 8; ++r) {
        int m = bm + wm + mt*16 + ((lane < 16) ? r : r + 8);
        int n = bn + wn + nt*16 + lm;
        size_t idx = (size_t)m * ldc + n;
        float v = acc[mt][nt][r];
        if (R1) v += R1[idx];
        if (R2) v += R2[idx];
        if (relu) v = fmaxf(v, 0.f);
        C[idx] = v;
    }
}

// ---------------------------------------------------------------------------
// Embedding gather: emb[n][:] = token_embed[tokens[n]][:]  (256 floats/token)
// ---------------------------------------------------------------------------
__global__ void embed_gather_kernel(const int* __restrict__ tokens,
                                    const float* __restrict__ token_embed,
                                    float* __restrict__ emb)
{
    int n = blockIdx.x;
    int tok = tokens[n];
    const float4* src = (const float4*)(token_embed + (size_t)tok * EDIM);
    float4* dst = (float4*)(emb + (size_t)n * EDIM);
    dst[threadIdx.x] = src[threadIdx.x];   // 64 threads * float4 = 256 floats
}

// ---------------------------------------------------------------------------
// Per-row exact K-th largest via 32-pass radix select on sortable keys,
// fused SDR binarization + hash-slot computation.
// ---------------------------------------------------------------------------
__global__ __launch_bounds__(256) void topk_sdr_kernel(
    const float* __restrict__ scores, float* __restrict__ sdr,
    int* __restrict__ slots, const int* __restrict__ hash_vec)
{
    __shared__ unsigned keys[SDIM];
    __shared__ int cnt;
    __shared__ int hsum;
    const int row = blockIdx.x, tid = threadIdx.x;
    const float* sr = scores + (size_t)row * SDIM;

    for (int i = tid; i < SDIM; i += 256) {
        union { float f; unsigned u; } a; a.f = sr[i];
        keys[i] = a.u ^ ((a.u >> 31) ? 0xFFFFFFFFu : 0x80000000u);  // order-preserving
    }
    if (tid == 0) { cnt = 0; hsum = 0; }
    __syncthreads();

    unsigned prefix = 0;
    int need = KACT;
    for (int bit = 31; bit >= 0; --bit) {
        unsigned hi = (bit == 31) ? 0u : (0xFFFFFFFFu << (bit + 1));
        int local = 0;
        for (int i = tid; i < SDIM; i += 256) {
            unsigned k = keys[i];
            if ((k & hi) == prefix && ((k >> bit) & 1u)) local++;
        }
        if (local) atomicAdd(&cnt, local);
        __syncthreads();
        int c = cnt;
        __syncthreads();
        if (tid == 0) cnt = 0;
        if (c >= need) prefix |= (1u << bit);
        else need -= c;
        __syncthreads();
    }
    // prefix == key of K-th largest; sdr = (score >= kth)
    int lh = 0;
    for (int i = tid; i < SDIM; i += 256) {
        int on = keys[i] >= prefix;
        sdr[(size_t)row * SDIM + i] = on ? 1.0f : 0.0f;
        if (on) lh += hash_vec[i];
    }
    if (lh) atomicAdd(&hsum, lh);
    __syncthreads();
    if (tid == 0) slots[row] = hsum & (CAP - 1);
}

// ---------------------------------------------------------------------------
// Row LayerNorm over EMB=768 (deterministic tree reductions)
// ---------------------------------------------------------------------------
__global__ __launch_bounds__(256) void layernorm_kernel(
    const float* __restrict__ X, const float* __restrict__ g,
    const float* __restrict__ b, float* __restrict__ Y)
{
    __shared__ float red[256];
    const int row = blockIdx.x, tid = threadIdx.x;
    const float* xr = X + (size_t)row * EMB;
    float s = 0.f;
    for (int i = tid; i < EMB; i += 256) s += xr[i];
    red[tid] = s; __syncthreads();
    for (int o = 128; o > 0; o >>= 1) { if (tid < o) red[tid] += red[tid + o]; __syncthreads(); }
    float mean = red[0] * (1.0f / EMB);
    __syncthreads();
    float v = 0.f;
    for (int i = tid; i < EMB; i += 256) { float d = xr[i] - mean; v += d * d; }
    red[tid] = v; __syncthreads();
    for (int o = 128; o > 0; o >>= 1) { if (tid < o) red[tid] += red[tid + o]; __syncthreads(); }
    float inv = rsqrtf(red[0] * (1.0f / EMB) + 1e-5f);
    for (int i = tid; i < EMB; i += 256)
        Y[(size_t)row * EMB + i] = (xr[i] - mean) * inv * g[i] + b[i];
}

// ---------------------------------------------------------------------------
// Per-row log-softmax NLL: row_loss[n] = -(logit[tgt] - max - log(sum exp))
// ---------------------------------------------------------------------------
__global__ __launch_bounds__(256) void loss_rows_kernel(
    const float* __restrict__ logits, const int* __restrict__ targets,
    float* __restrict__ row_loss)
{
    __shared__ float red[256];
    const int row = blockIdx.x, tid = threadIdx.x;
    const float* lr = logits + (size_t)row * VOCAB;
    float mx = -INFINITY;
    for (int i = tid; i < VOCAB; i += 256) mx = fmaxf(mx, lr[i]);
    red[tid] = mx; __syncthreads();
    for (int o = 128; o > 0; o >>= 1) { if (tid < o) red[tid] = fmaxf(red[tid], red[tid + o]); __syncthreads(); }
    mx = red[0];
    __syncthreads();
    float se = 0.f;
    for (int i = tid; i < VOCAB; i += 256) se += expf(lr[i] - mx);
    red[tid] = se; __syncthreads();
    for (int o = 128; o > 0; o >>= 1) { if (tid < o) red[tid] += red[tid + o]; __syncthreads(); }
    if (tid == 0) {
        int t = targets[row];
        row_loss[row] = -(lr[t] - mx - logf(red[0]));
    }
}

__global__ __launch_bounds__(256) void loss_reduce_kernel(
    const float* __restrict__ row_loss, float* __restrict__ out_loss)
{
    __shared__ float red[256];
    int tid = threadIdx.x;
    float s = 0.f;
    for (int i = tid; i < NTOK; i += 256) s += row_loss[i];
    red[tid] = s; __syncthreads();
    for (int o = 128; o > 0; o >>= 1) { if (tid < o) red[tid] += red[tid + o]; __syncthreads(); }
    if (tid == 0) out_loss[0] = red[0] / (float)NTOK;
}

// ---------------------------------------------------------------------------
// Engram scatter: mem_out[slots[n]] += xf[n]  (global_atomic_add_f32)
// ---------------------------------------------------------------------------
__global__ void mem_scatter_kernel(const float* __restrict__ xf,
                                   const int* __restrict__ slots,
                                   float* __restrict__ mem_out)
{
    int n = blockIdx.x;
    int slot = slots[n];
    float* dst = mem_out + (size_t)slot * EMB;
    const float* src = xf + (size_t)n * EMB;
    for (int e = threadIdx.x; e < EMB; e += blockDim.x)
        atomicAdd(dst + e, src[e]);
}

// ---------------------------------------------------------------------------
static inline void launch_gemm(const float* A, int lda, const float* B, int ldb,
                               float* C, int ldc, const float* R1, const float* R2,
                               int M, int N, int Kd, int relu, hipStream_t s)
{
    dim3 grid(N / 128, M / 128);
    gemm_bf16_wmma<<<grid, 256, 0, s>>>(A, lda, B, ldb, C, ldc, R1, R2, Kd, relu);
}

extern "C" void kernel_launch(void* const* d_in, const int* in_sizes, int n_in,
                              void* d_out, int out_size, void* d_ws, size_t ws_size,
                              hipStream_t stream)
{
    (void)in_sizes; (void)n_in; (void)out_size; (void)ws_size;

    const int*   tokens      = (const int*)  d_in[0];
    const int*   targets     = (const int*)  d_in[1];
    const float* token_embed = (const float*)d_in[2];
    const float* enc_proj    = (const float*)d_in[3];
    const float* W_in        = (const float*)d_in[4];
    const float* W1          = (const float*)d_in[5];
    const float* W2          = (const float*)d_in[6];
    const float* ln_g        = (const float*)d_in[7];
    const float* ln_b        = (const float*)d_in[8];
    const float* lnf_g       = (const float*)d_in[9];
    const float* lnf_b       = (const float*)d_in[10];
    const float* lm_head     = (const float*)d_in[11];
    const float* e2s_W       = (const float*)d_in[12];
    const float* memory      = (const float*)d_in[13];
    const int*   hash_vec    = (const int*)  d_in[14];

    float* out_logits = (float*)d_out;                       // [2048, 32000]
    float* out_loss   = out_logits + (size_t)NTOK * VOCAB;   // [1]
    float* out_mem    = out_loss + 1;                        // [65536, 768]

    char* wp = (char*)d_ws;
    auto carve = [&](size_t bytes) -> void* {
        void* p = (void*)wp; wp += (bytes + 255) & ~(size_t)255; return p;
    };
    float* emb      = (float*)carve((size_t)NTOK * EDIM * 4);
    float* scores   = (float*)carve((size_t)NTOK * SDIM * 4);
    float* sdr      = (float*)carve((size_t)NTOK * SDIM * 4);
    float* xs       = (float*)carve((size_t)NTOK * SDIM * 4);
    float* h        = (float*)carve((size_t)NTOK * EMB  * 4);
    float* aln      = (float*)carve((size_t)NTOK * EMB  * 4);
    float* hm       = (float*)carve((size_t)NTOK * HID  * 4);
    float* x0       = (float*)carve((size_t)NTOK * EMB  * 4);
    float* x1       = (float*)carve((size_t)NTOK * EMB  * 4);
    float* xf       = (float*)carve((size_t)NTOK * EMB  * 4);
    float* row_loss = (float*)carve((size_t)NTOK * 4);
    int*   slots    = (int*)  carve((size_t)NTOK * 4);

    // 1) encode: gather + scores GEMM + topK/SDR/hash
    embed_gather_kernel<<<NTOK, 64, 0, stream>>>(tokens, token_embed, emb);
    launch_gemm(emb, EDIM, enc_proj, SDIM, scores, SDIM,
                nullptr, nullptr, NTOK, SDIM, EDIM, 0, stream);
    topk_sdr_kernel<<<NTOK, 256, 0, stream>>>(scores, sdr, slots, hash_vec);

    // 2) layers
    float* xcur = x0;
    float* xnext = x1;
    for (int l = 0; l < NLAYER; ++l) {
        const float* s_in = (l == 0) ? sdr : xs;
        if (l > 0)   // xs = x @ e2s_W + sdr
            launch_gemm(xcur, EMB, e2s_W, SDIM, xs, SDIM,
                        sdr, nullptr, NTOK, SDIM, EMB, 0, stream);
        // h = s_in @ W_in[l]
        launch_gemm(s_in, SDIM, W_in + (size_t)l * SDIM * EMB, EMB, h, EMB,
                    nullptr, nullptr, NTOK, EMB, SDIM, 0, stream);
        // a = LN(h)
        layernorm_kernel<<<NTOK, 256, 0, stream>>>(
            h, ln_g + (size_t)l * EMB, ln_b + (size_t)l * EMB, aln);
        // hm = relu(a @ W1[l])
        launch_gemm(aln, EMB, W1 + (size_t)l * EMB * HID, HID, hm, HID,
                    nullptr, nullptr, NTOK, HID, EMB, 1, stream);
        // x_new = hm @ W2[l] + h (+ x for layers > 0)
        launch_gemm(hm, HID, W2 + (size_t)l * HID * EMB, EMB, xnext, EMB,
                    h, (l == 0) ? nullptr : xcur, NTOK, EMB, HID, 0, stream);
        float* t = xcur; xcur = xnext; xnext = t;
    }

    // 3) final LN
    layernorm_kernel<<<NTOK, 256, 0, stream>>>(xcur, lnf_g, lnf_b, xf);

    // 4) logits + loss
    launch_gemm(xf, EMB, lm_head, VOCAB, out_logits, VOCAB,
                nullptr, nullptr, NTOK, VOCAB, EMB, 0, stream);
    loss_rows_kernel<<<NTOK, 256, 0, stream>>>(out_logits, targets, row_loss);
    loss_reduce_kernel<<<1, 256, 0, stream>>>(row_loss, out_loss);

    // 5) engram memory: copy + scatter-add
    hipMemcpyAsync(out_mem, memory, (size_t)CAP * EMB * sizeof(float),
                   hipMemcpyDeviceToDevice, stream);
    mem_scatter_kernel<<<NTOK, 256, 0, stream>>>(xf, slots, out_mem);
}
